// GroupSplitter_66099546685673
// MI455X (gfx1250) — matrix operand order repaired and checked by reading, not compiled, not run on hardware
//
#include <hip/hip_runtime.h>
#include <hip/hip_bf16.h>

typedef __attribute__((ext_vector_type(16))) __bf16 v16bf;
typedef __attribute__((ext_vector_type(8)))  __bf16 v8bf;
typedef __attribute__((ext_vector_type(8)))  float  v8f;

#define HID 128
#define NH 4
#define CH 32
#define ED 4
#define NL 3

// ---------------- utility ----------------
__global__ void fill_kernel(float* __restrict__ p, float v, int n) {
    int i = blockIdx.x * blockDim.x + threadIdx.x;
    if (i < n) p[i] = v;
}

__device__ __forceinline__ void atomicMaxF(float* addr, float val) {
    int old = __float_as_int(*addr);
    while (__int_as_float(old) < val) {
        int prev = atomicCAS((int*)addr, old, __float_as_int(val));
        if (prev == old) break;
        old = prev;
    }
}

// Convert 4 weight tensors [L,128,128] (K-major) -> bf16 transposed [sel][l][n][k]
__global__ void prep_weights(const float* __restrict__ Wq, const float* __restrict__ Wk,
                             const float* __restrict__ Wv, const float* __restrict__ Ws,
                             __bf16* __restrict__ Wt) {
    int idx = blockIdx.x * blockDim.x + threadIdx.x;   // 4*3*16384 threads
    int sel = idx / (NL * HID * HID);
    int r   = idx % (NL * HID * HID);
    int l   = r / (HID * HID);
    int j   = r % (HID * HID);
    int n   = j / HID, k = j % HID;
    const float* W = (sel == 0) ? Wq : (sel == 1) ? Wk : (sel == 2) ? Wv : Ws;
    Wt[idx] = (__bf16)W[l * HID * HID + k * HID + n];
}

// h = x @ Win + bin  (K = 5), also emits bf16 activation copy
__global__ void input_proj(const float* __restrict__ x, const float* __restrict__ Win,
                           const float* __restrict__ bin, float* __restrict__ h,
                           __bf16* __restrict__ hA, int N) {
    int idx = blockIdx.x * blockDim.x + threadIdx.x;
    if (idx >= N * HID) return;
    int n = idx >> 7, c = idx & 127;
    float acc = bin[c];
#pragma unroll
    for (int d = 0; d < 5; d++) acc += x[n * 5 + d] * Win[d * HID + c];
    h[idx]  = acc;
    hA[idx] = (__bf16)acc;
}

// ------------- fused WMMA bf16 GEMM x4: {q,k,v,xr} = A @ {Wq,Wk,Wv,Ws}^T + bias -------------
// A: bf16 row-major [row][k].  Wt*: bf16 [n][k] (pre-transposed).  outs: f32 [N][128].
// Block = 256 thr = 8 waves; block handles 16 rows; wave w handles cols [16w,16w+16).
// A tile staged once in LDS; A fragments register-resident across all 4 matrices.
__global__ __launch_bounds__(256) void gemm4_bf16_128(
        const __bf16* __restrict__ A,
        const __bf16* __restrict__ WtQ, const __bf16* __restrict__ WtK,
        const __bf16* __restrict__ WtV, const __bf16* __restrict__ WtS,
        const float* __restrict__ bq, const float* __restrict__ bk,
        const float* __restrict__ bv, const float* __restrict__ bs,
        float* __restrict__ oq, float* __restrict__ ok,
        float* __restrict__ ov, float* __restrict__ os) {
    __shared__ __bf16 sA[16 * HID];                       // 4 KB A tile
    int tid = threadIdx.x;
    int rowBase = blockIdx.x * 16;
    // cooperative stage: 256 threads x 16B = 4 KB, single coalesced b128 each
    ((uint4*)sA)[tid] = ((const uint4*)(A + (size_t)rowBase * HID))[tid];
    __syncthreads();

    int lane = tid & 31;
    int wave = tid >> 5;
    int colBase = wave * 16;
    bool hi = (lane & 16) != 0;
    const __bf16* arow = sA + (lane & 15) * HID;

    // Build A fragments for all 4 K-steps once (ISA 7.12.2 16-bit A 16x32 layout):
    // lane<16 holds K=k0..k0+7 / k0+16..k0+23; lane>=16 holds K=k0+8..15 / k0+24..31
    v16bf afr[4];
#pragma unroll
    for (int ks = 0; ks < 4; ks++) {
        int ka = ks * 32 + (hi ? 8 : 0);
        v8bf a0 = *(const v8bf*)(arow + ka);
        v8bf a1 = *(const v8bf*)(arow + ka + 16);
#pragma unroll
        for (int i = 0; i < 8; i++) { afr[ks][i] = a0[i]; afr[ks][i + 8] = a1[i]; }
    }

    const __bf16* W[4]    = {WtQ, WtK, WtV, WtS};
    const float*  bias[4] = {bq, bk, bv, bs};
    float*        out[4]  = {oq, ok, ov, os};
    int nCol = colBase + (lane & 15);
    int mB   = hi ? 8 : 0;

#pragma unroll
    for (int m = 0; m < 4; m++) {
        const __bf16* brow = W[m] + (size_t)nCol * HID;
        v8f acc = {};
#pragma unroll
        for (int ks = 0; ks < 4; ks++) {
            // B fragment (32x16): lane<16 -> K=k0..k0+7 / k0+16..k0+23, lane>=16 -> K=k0+8..k0+15 / k0+24..k0+31
            int kb = ks * 32 + (hi ? 8 : 0);
            v8bf b0 = *(const v8bf*)(brow + kb);
            v8bf b1 = *(const v8bf*)(brow + kb + 16);
            v16bf bfr;
#pragma unroll
            for (int i = 0; i < 8; i++) { bfr[i] = b0[i]; bfr[i + 8] = b1[i]; }
            acc = __builtin_amdgcn_wmma_f32_16x16x32_bf16(false, afr[ks], false, bfr,
                                                          (short)0, acc, false, false);
        }
        float bvl = bias[m][nCol];
#pragma unroll
        for (int v = 0; v < 8; v++)
            out[m][(size_t)(rowBase + mB + v) * HID + nCol] = acc[v] + bvl;
    }
}

// ------------- edge pass 1: attention logits + segment max -------------
__global__ __launch_bounds__(256) void edge_logits(const long long* __restrict__ src,
                                                   const long long* __restrict__ dst,
                                                   const float* __restrict__ ea,
                                                   const float* __restrict__ We,
                                                   const float* __restrict__ q,
                                                   const float* __restrict__ k,
                                                   float* __restrict__ alpha,
                                                   float* __restrict__ amax, int E4) {
    __shared__ float sWe[ED * HID];
    int tid = threadIdx.x;
    sWe[tid] = We[tid];
    sWe[tid + 256] = We[tid + 256];
    __syncthreads();
    int idx = blockIdx.x * 256 + tid;
    if (idx >= E4) return;
    int e = idx >> 2, h = idx & 3;
    long long s = src[e], d = dst[e];
    float a0 = ea[e * 4 + 0], a1 = ea[e * 4 + 1], a2 = ea[e * 4 + 2], a3 = ea[e * 4 + 3];
    const float* qp = q + (size_t)d * HID + h * CH;
    const float* kp = k + (size_t)s * HID + h * CH;
    const float* w  = sWe + h * CH;
    float acc = 0.f;
#pragma unroll
    for (int c = 0; c < CH; c++) {
        float ec = a0 * w[c] + a1 * w[HID + c] + a2 * w[2 * HID + c] + a3 * w[3 * HID + c];
        acc += qp[c] * (kp[c] + ec);
    }
    acc *= 0.17677669529663687f;   // 1/sqrt(32)
    alpha[idx] = acc;
    atomicMaxF(&amax[d * NH + h], acc);
}

// ------------- edge pass 2: scatter exp numerator + denominator -------------
__global__ __launch_bounds__(256) void edge_scatter(const long long* __restrict__ src,
                                                    const long long* __restrict__ dst,
                                                    const float* __restrict__ ea,
                                                    const float* __restrict__ We,
                                                    const float* __restrict__ v,
                                                    const float* __restrict__ alpha,
                                                    const float* __restrict__ amax,
                                                    float* __restrict__ den,
                                                    float* __restrict__ agg, int E4) {
    __shared__ float sWe[ED * HID];
    int tid = threadIdx.x;
    sWe[tid] = We[tid];
    sWe[tid + 256] = We[tid + 256];
    __syncthreads();
    int idx = blockIdx.x * 256 + tid;
    if (idx >= E4) return;
    int e = idx >> 2, h = idx & 3;
    long long s = src[e], d = dst[e];
    float ex = __expf(alpha[idx] - amax[d * NH + h]);
    atomicAdd(&den[d * NH + h], ex);
    float a0 = ea[e * 4 + 0], a1 = ea[e * 4 + 1], a2 = ea[e * 4 + 2], a3 = ea[e * 4 + 3];
    const float* vp = v + (size_t)s * HID + h * CH;
    float* op = agg + (size_t)d * HID + h * CH;
    const float* w = sWe + h * CH;
#pragma unroll
    for (int c = 0; c < CH; c++) {
        float ec = a0 * w[c] + a1 * w[HID + c] + a2 * w[2 * HID + c] + a3 * w[3 * HID + c];
        atomicAdd(&op[c], ex * (vp[c] + ec));
    }
}

// ------------- node combine: softmax-normalize + beta gate + ReLU + LayerNorm -------------
// One wave32 per node; each lane owns 4 contiguous channels.
__global__ __launch_bounds__(256) void node_combine(const float* __restrict__ agg,
                                                    const float* __restrict__ den,
                                                    const float* __restrict__ xr,
                                                    const float* __restrict__ Wb,
                                                    const float* __restrict__ lng,
                                                    const float* __restrict__ lnb,
                                                    float* __restrict__ h,
                                                    __bf16* __restrict__ hA, int N) {
    int lane = threadIdx.x & 31;
    int node = blockIdx.x * 8 + (threadIdx.x >> 5);
    if (node >= N) return;
    int head = lane >> 3;                    // lane*4..lane*4+3 all in one head
    float inv = 1.0f / (den[node * NH + head] + 1e-16f);
    size_t base = (size_t)node * HID + lane * 4;
    float o[4], xv[4];
    float dotp = 0.f;
#pragma unroll
    for (int j = 0; j < 4; j++) {
        int ch = lane * 4 + j;
        o[j]  = agg[base + j] * inv;
        xv[j] = xr[base + j];
        dotp += o[j] * Wb[ch] + xv[j] * Wb[HID + ch] + (o[j] - xv[j]) * Wb[2 * HID + ch];
    }
#pragma unroll
    for (int off = 16; off >= 1; off >>= 1) dotp += __shfl_xor(dotp, off, 32);
    float beta = 1.0f / (1.0f + __expf(-dotp));
    float hn[4]; float ls = 0.f;
#pragma unroll
    for (int j = 0; j < 4; j++) {
        hn[j] = fmaxf(beta * xv[j] + (1.0f - beta) * o[j], 0.0f);  // relu
        ls += hn[j];
    }
#pragma unroll
    for (int off = 16; off >= 1; off >>= 1) ls += __shfl_xor(ls, off, 32);
    float mean = ls * (1.0f / 128.0f);
    float vs = 0.f;
#pragma unroll
    for (int j = 0; j < 4; j++) { float dd = hn[j] - mean; vs += dd * dd; }
#pragma unroll
    for (int off = 16; off >= 1; off >>= 1) vs += __shfl_xor(vs, off, 32);
    float r = rsqrtf(vs * (1.0f / 128.0f) + 1e-5f);
#pragma unroll
    for (int j = 0; j < 4; j++) {
        int ch = lane * 4 + j;
        float y = (hn[j] - mean) * r * lng[ch] + lnb[ch];
        h[base + j]  = y;
        hA[base + j] = (__bf16)y;
    }
}

// ------------- final classifier: out = h @ Wh + bh  (128 -> 3) -------------
__global__ void classify(const float* __restrict__ h, const float* __restrict__ Wh,
                         const float* __restrict__ bh, float* __restrict__ out, int N) {
    int n = blockIdx.x * blockDim.x + threadIdx.x;
    if (n >= N) return;
    float a0 = bh[0], a1 = bh[1], a2 = bh[2];
    const float* hp = h + (size_t)n * HID;
#pragma unroll 8
    for (int c = 0; c < HID; c++) {
        float t = hp[c];
        a0 += t * Wh[c * 3 + 0];
        a1 += t * Wh[c * 3 + 1];
        a2 += t * Wh[c * 3 + 2];
    }
    out[n * 3 + 0] = a0; out[n * 3 + 1] = a1; out[n * 3 + 2] = a2;
}

extern "C" void kernel_launch(void* const* d_in, const int* in_sizes, int n_in,
                              void* d_out, int out_size, void* d_ws, size_t ws_size,
                              hipStream_t stream) {
    const float*     x    = (const float*)d_in[0];
    const long long* ei   = (const long long*)d_in[1];
    const float*     ea   = (const float*)d_in[2];
    const float*     Win  = (const float*)d_in[3];
    const float*     bin  = (const float*)d_in[4];
    const float*     Wq   = (const float*)d_in[5];
    const float*     bq   = (const float*)d_in[6];
    const float*     Wk   = (const float*)d_in[7];
    const float*     bk   = (const float*)d_in[8];
    const float*     Wv   = (const float*)d_in[9];
    const float*     bv   = (const float*)d_in[10];
    const float*     We   = (const float*)d_in[11];
    const float*     Ws   = (const float*)d_in[12];
    const float*     bs   = (const float*)d_in[13];
    const float*     Wb   = (const float*)d_in[14];
    const float*     lng  = (const float*)d_in[15];
    const float*     lnb  = (const float*)d_in[16];
    const float*     Wh   = (const float*)d_in[17];
    const float*     bh   = (const float*)d_in[18];

    const int N = in_sizes[0] / 5;      // 50000
    const int E = in_sizes[1] / 2;      // 800000
    const long long* srcI = ei;
    const long long* dstI = ei + E;

    // workspace carving (256B aligned)
    char* wsp = (char*)d_ws;
    size_t off = 0;
    auto alloc = [&](size_t bytes) -> char* {
        char* p = wsp + off;
        off = (off + bytes + 255) & ~(size_t)255;
        return p;
    };
    __bf16* hA  = (__bf16*)alloc((size_t)N * HID * 2);
    float*  hF  = (float*)alloc((size_t)N * HID * 4);
    float*  qB  = (float*)alloc((size_t)N * HID * 4);
    float*  kB  = (float*)alloc((size_t)N * HID * 4);
    float*  vB  = (float*)alloc((size_t)N * HID * 4);
    float*  xrB = (float*)alloc((size_t)N * HID * 4);
    float*  agg = (float*)alloc((size_t)N * HID * 4);
    float*  alp = (float*)alloc((size_t)E * NH * 4);
    float*  amx = (float*)alloc((size_t)N * NH * 4);
    float*  den = (float*)alloc((size_t)N * NH * 4);
    __bf16* Wt  = (__bf16*)alloc((size_t)4 * NL * HID * HID * 2);

    // one-time (per launch) weight conversion to transposed bf16
    prep_weights<<<(4 * NL * HID * HID + 255) / 256, 256, 0, stream>>>(Wq, Wk, Wv, Ws, Wt);
    input_proj<<<(N * HID + 255) / 256, 256, 0, stream>>>(x, Win, bin, hF, hA, N);

    const int nb16 = (N + 15) / 16;     // 3125, exact
    const int E4   = E * NH;
    for (int l = 0; l < NL; l++) {
        const __bf16* WtQ = Wt + (size_t)(0 * NL + l) * HID * HID;
        const __bf16* WtK = Wt + (size_t)(1 * NL + l) * HID * HID;
        const __bf16* WtV = Wt + (size_t)(2 * NL + l) * HID * HID;
        const __bf16* WtS = Wt + (size_t)(3 * NL + l) * HID * HID;
        gemm4_bf16_128<<<nb16, 256, 0, stream>>>(hA, WtQ, WtK, WtV, WtS,
                                                 bq + l * HID, bk + l * HID,
                                                 bv + l * HID, bs + l * HID,
                                                 qB, kB, vB, xrB);

        fill_kernel<<<(N * NH + 255) / 256, 256, 0, stream>>>(amx, -3.4e38f, N * NH);
        fill_kernel<<<(N * NH + 255) / 256, 256, 0, stream>>>(den, 0.0f, N * NH);
        fill_kernel<<<(N * HID + 255) / 256, 256, 0, stream>>>(agg, 0.0f, N * HID);

        edge_logits<<<(E4 + 255) / 256, 256, 0, stream>>>(srcI, dstI, ea, We + l * ED * HID,
                                                          qB, kB, alp, amx, E4);
        edge_scatter<<<(E4 + 255) / 256, 256, 0, stream>>>(srcI, dstI, ea, We + l * ED * HID,
                                                           vB, alp, amx, den, agg, E4);
        node_combine<<<(N + 7) / 8, 256, 0, stream>>>(agg, den, xrB, Wb + l * 3 * HID,
                                                      lng + l * HID, lnb + l * HID, hF, hA, N);
    }
    classify<<<(N + 255) / 256, 256, 0, stream>>>(hF, Wh, bh, (float*)d_out, N);
}
